// CausalSelfAttention_6743098654971
// MI455X (gfx1250) — matrix operand order repaired and checked by reading, not compiled, not run on hardware
//
#include <hip/hip_runtime.h>
#include <hip/hip_bf16.h>

// ---------------------------------------------------------------------------
// CDNA5 (gfx1250) causal self-attention, WMMA f16 path, f32 accumulation.
//   N_EMBED=1024, N_HEAD=16, HD=64, B=2, S=2048  ->  rows M = 4096
// ---------------------------------------------------------------------------

typedef __attribute__((ext_vector_type(16))) _Float16 v16h;
typedef __attribute__((ext_vector_type(8)))  _Float16 v8h;
typedef __attribute__((ext_vector_type(8)))  float    v8f;

#define D_EMB   1024
#define D_QKV   3072
#define N_HEADS 16
#define HD      64
#define SEQ     2048
#define NBATCH  2
#define MROWS   (NBATCH * SEQ)   // 4096

__device__ __forceinline__ v8f wmma_f16(v16h a, v16h b, v8f c) {
    // 8 args: (neg_a, A, neg_b, B, c_mod, C, reuse_a, reuse_b)
    return __builtin_amdgcn_wmma_f32_16x16x32_f16(false, a, false, b,
                                                  (short)0, c, false, false);
}

// A-matrix 16x32 f16 frag from row-major [*, ld] f16:
//   lane&15 selects M row; elements i: K = k0 + 8*(lane>>4) + (i&7) + 16*(i>>3)
__device__ __forceinline__ v16h load_fragA(const _Float16* __restrict__ base,
                                           int ld, int row, int k0, int lane) {
    const _Float16* p = base + (size_t)row * ld + k0 + 8 * (lane >> 4);
    union { v16h v; v8h h[2]; } u;
    u.h[0] = *(const v8h*)(p);
    u.h[1] = *(const v8h*)(p + 16);
    return u.v;
}

// B-matrix 32x16 f16 frag from [N,K] row-major (B stored transposed):
//   lane&15 selects N col; elements i: K = k0 + 8*(lane>>4) + (i&7) + 16*(i>>3)
__device__ __forceinline__ v16h load_fragB(const _Float16* __restrict__ base,
                                           int ld, int nrow, int k0, int lane) {
    const _Float16* p = base + (size_t)nrow * ld + k0 + 8 * (lane >> 4);
    union { v16h v; v8h h[2]; } u;
    u.h[0] = *(const v8h*)(p);
    u.h[1] = *(const v8h*)(p + 16);
    return u.v;
}

// ---------------------------------------------------------------------------
// Pre-passes
// ---------------------------------------------------------------------------
__global__ void cvt_f32_f16(const float* __restrict__ in,
                            _Float16* __restrict__ out, int n) {
    int i = blockIdx.x * blockDim.x + threadIdx.x;
    if (i < n) out[i] = (_Float16)in[i];
}

// in[rows][cols] f32 -> out[cols][rows] f16   (weights become [N,K] row-major)
__global__ void transpose_f32_f16(const float* __restrict__ in,
                                  _Float16* __restrict__ out, int rows, int cols) {
    int i = blockIdx.x * blockDim.x + threadIdx.x;
    if (i < rows * cols) {
        int r = i / cols, c = i % cols;
        out[(size_t)c * rows + r] = (_Float16)in[i];
    }
}

// ---------------------------------------------------------------------------
// QKV GEMM: Xh[4096,1024] @ WqkvT'[3072,1024] + b_qkv
//   Q scattered row-major per head (scaled by 1/8), K row-major per head,
//   V transposed per head ([hd, s]).  One wave -> 64x64 tile.
// ---------------------------------------------------------------------------
__global__ __launch_bounds__(32)
void qkv_gemm(const _Float16* __restrict__ Xh, const _Float16* __restrict__ WT,
              const float* __restrict__ bias,
              _Float16* __restrict__ Qh, _Float16* __restrict__ Kh,
              _Float16* __restrict__ Vt) {
    const int lane = threadIdx.x & 31;
    const int nl = lane & 15, g = lane >> 4;
    const int m0 = blockIdx.x * 64;
    const int n0 = blockIdx.y * 64;

    v8f acc[4][4] = {};
    for (int k0 = 0; k0 < D_EMB; k0 += 32) {
        v16h a[4], b[4];
        #pragma unroll
        for (int i = 0; i < 4; ++i)
            a[i] = load_fragA(Xh, D_EMB, m0 + i * 16 + nl, k0, lane);
        #pragma unroll
        for (int j = 0; j < 4; ++j)
            b[j] = load_fragB(WT, D_EMB, n0 + j * 16 + nl, k0, lane);
        #pragma unroll
        for (int i = 0; i < 4; ++i)
            #pragma unroll
            for (int j = 0; j < 4; ++j)
                acc[i][j] = wmma_f16(a[i], b[j], acc[i][j]);
    }

    #pragma unroll
    for (int j = 0; j < 4; ++j) {
        const int n = n0 + j * 16 + nl;
        const float bv = bias[n];
        const int f = n & (D_EMB - 1);
        const int h = f >> 6, d = f & (HD - 1);
        #pragma unroll
        for (int i = 0; i < 4; ++i) {
            #pragma unroll
            for (int r = 0; r < 8; ++r) {
                const int m = m0 + i * 16 + r + 8 * g;
                const int bb = m >> 11, s = m & (SEQ - 1);
                const size_t bh = (size_t)(bb * N_HEADS + h);
                const float c = acc[i][j][r] + bv;
                if (n < D_EMB)            // Q: fold in softmax scale 1/sqrt(64)
                    Qh[(bh * SEQ + s) * HD + d] = (_Float16)(c * 0.125f);
                else if (n < 2 * D_EMB)   // K row-major == [N,K] for Q@K^T
                    Kh[(bh * SEQ + s) * HD + d] = (_Float16)c;
                else                      // V transposed == [N,K] for P@V
                    Vt[(bh * HD + d) * SEQ + s] = (_Float16)c;
            }
        }
    }
}

// ---------------------------------------------------------------------------
// Flash attention: one wave per (batch*head, 32-query tile).
//   64-key blocks, online softmax, causal mask on diagonal blocks.
// ---------------------------------------------------------------------------
__global__ __launch_bounds__(32)
void attn_fwd(const _Float16* __restrict__ Qh, const _Float16* __restrict__ Kh,
              const _Float16* __restrict__ Vt, _Float16* __restrict__ attnH) {
    __shared__ _Float16 Pl[32][72];   // padded: 16B-aligned rows, conflict-light

    const int lane = threadIdx.x & 31;
    const int nl = lane & 15, g = lane >> 4;
    const int q0 = blockIdx.x * 32;
    const int bh = blockIdx.y;
    const _Float16* Qp = Qh + (size_t)bh * SEQ * HD;
    const _Float16* Kp = Kh + (size_t)bh * SEQ * HD;
    const _Float16* Vp = Vt + (size_t)bh * HD * SEQ;

    v16h qf[2][2];
    #pragma unroll
    for (int mf = 0; mf < 2; ++mf)
        #pragma unroll
        for (int kf = 0; kf < 2; ++kf)
            qf[mf][kf] = load_fragA(Qp, HD, q0 + mf * 16 + nl, kf * 32, lane);

    v8f o[2][4] = {};
    float mstat[2][8], lstat[2][8];
    #pragma unroll
    for (int mf = 0; mf < 2; ++mf)
        #pragma unroll
        for (int r = 0; r < 8; ++r) { mstat[mf][r] = -1e30f; lstat[mf][r] = 0.f; }

    const int nblocks = (q0 + 32 + 63) / 64;
    for (int kb = 0; kb < nblocks; ++kb) {
        const int k0 = kb * 64;

        // ---- S = (Q/8) @ K^T  (64-key block) ----
        v8f s[2][4];
        #pragma unroll
        for (int nf = 0; nf < 4; ++nf) {
            v16h bk0 = load_fragB(Kp, HD, k0 + nf * 16 + nl, 0, lane);
            v16h bk1 = load_fragB(Kp, HD, k0 + nf * 16 + nl, 32, lane);
            #pragma unroll
            for (int mf = 0; mf < 2; ++mf) {
                v8f z = {};
                v8f t = wmma_f16(qf[mf][0], bk0, z);
                s[mf][nf] = wmma_f16(qf[mf][1], bk1, t);
            }
        }

        // ---- causal mask (diagonal blocks only) ----
        if (k0 + 63 > q0) {
            #pragma unroll
            for (int mf = 0; mf < 2; ++mf)
                #pragma unroll
                for (int nf = 0; nf < 4; ++nf) {
                    const int kpos = k0 + nf * 16 + nl;
                    #pragma unroll
                    for (int r = 0; r < 8; ++r) {
                        const int qpos = q0 + mf * 16 + r + 8 * g;
                        if (kpos > qpos) s[mf][nf][r] = -1e30f;
                    }
                }
        }

        // ---- online softmax (row stats across 16-lane halves) ----
        #pragma unroll
        for (int mf = 0; mf < 2; ++mf) {
            #pragma unroll
            for (int r = 0; r < 8; ++r) {
                float v = fmaxf(fmaxf(s[mf][0][r], s[mf][1][r]),
                                fmaxf(s[mf][2][r], s[mf][3][r]));
                #pragma unroll
                for (int off = 8; off >= 1; off >>= 1)
                    v = fmaxf(v, __shfl_xor(v, off, 32));
                const float mnew  = fmaxf(mstat[mf][r], v);
                const float alpha = __expf(mstat[mf][r] - mnew);
                mstat[mf][r] = mnew;
                float sum = 0.f;
                #pragma unroll
                for (int nf = 0; nf < 4; ++nf) {
                    const float p = __expf(s[mf][nf][r] - mnew);
                    s[mf][nf][r] = p;
                    sum += p;
                }
                #pragma unroll
                for (int off = 8; off >= 1; off >>= 1)
                    sum += __shfl_xor(sum, off, 32);
                lstat[mf][r] = lstat[mf][r] * alpha + sum;
                #pragma unroll
                for (int nf = 0; nf < 4; ++nf)
                    o[mf][nf][r] *= alpha;
            }
        }

        // ---- P (C-layout f32) -> LDS f16 row-major -> A-layout frags ----
        __syncthreads();
        #pragma unroll
        for (int mf = 0; mf < 2; ++mf)
            #pragma unroll
            for (int nf = 0; nf < 4; ++nf)
                #pragma unroll
                for (int r = 0; r < 8; ++r)
                    Pl[mf * 16 + r + 8 * g][nf * 16 + nl] = (_Float16)s[mf][nf][r];
        __syncthreads();

        v16h pf[2][2];
        #pragma unroll
        for (int mf = 0; mf < 2; ++mf)
            #pragma unroll
            for (int kf = 0; kf < 2; ++kf) {
                const _Float16* p = &Pl[mf * 16 + nl][kf * 32 + 8 * g];
                union { v16h v; v8h h[2]; } u;
                u.h[0] = *(const v8h*)(p);
                u.h[1] = *(const v8h*)(p + 16);
                pf[mf][kf] = u.v;
            }

        // ---- O += P @ V  (V stored [hd, s] so B-frags are contiguous) ----
        #pragma unroll
        for (int nf = 0; nf < 4; ++nf) {
            v16h bv0 = load_fragB(Vp, SEQ, nf * 16 + nl, k0, lane);
            v16h bv1 = load_fragB(Vp, SEQ, nf * 16 + nl, k0 + 32, lane);
            #pragma unroll
            for (int mf = 0; mf < 2; ++mf) {
                o[mf][nf] = wmma_f16(pf[mf][0], bv0, o[mf][nf]);
                o[mf][nf] = wmma_f16(pf[mf][1], bv1, o[mf][nf]);
            }
        }
        __syncthreads();
    }

    // ---- normalize + merge heads: attnH[b*S + q][h*64 + d] f16 ----
    const int bb = bh >> 4, h = bh & (N_HEADS - 1);
    #pragma unroll
    for (int mf = 0; mf < 2; ++mf)
        #pragma unroll
        for (int nf = 0; nf < 4; ++nf)
            #pragma unroll
            for (int r = 0; r < 8; ++r) {
                const int grow = bb * SEQ + q0 + mf * 16 + r + 8 * g;
                const int col  = h * HD + nf * 16 + nl;
                attnH[(size_t)grow * D_EMB + col] =
                    (_Float16)(o[mf][nf][r] / lstat[mf][r]);
            }
}

// ---------------------------------------------------------------------------
// Out projection: attnH[4096,1024] @ WoutT'[1024,1024] + b_out -> f32 d_out
// ---------------------------------------------------------------------------
__global__ __launch_bounds__(32)
void out_gemm(const _Float16* __restrict__ A, const _Float16* __restrict__ WT,
              const float* __restrict__ bias, float* __restrict__ out) {
    const int lane = threadIdx.x & 31;
    const int nl = lane & 15, g = lane >> 4;
    const int m0 = blockIdx.x * 64;
    const int n0 = blockIdx.y * 64;

    v8f acc[4][4] = {};
    for (int k0 = 0; k0 < D_EMB; k0 += 32) {
        v16h a[4], b[4];
        #pragma unroll
        for (int i = 0; i < 4; ++i)
            a[i] = load_fragA(A, D_EMB, m0 + i * 16 + nl, k0, lane);
        #pragma unroll
        for (int j = 0; j < 4; ++j)
            b[j] = load_fragB(WT, D_EMB, n0 + j * 16 + nl, k0, lane);
        #pragma unroll
        for (int i = 0; i < 4; ++i)
            #pragma unroll
            for (int j = 0; j < 4; ++j)
                acc[i][j] = wmma_f16(a[i], b[j], acc[i][j]);
    }

    #pragma unroll
    for (int j = 0; j < 4; ++j) {
        const int n = n0 + j * 16 + nl;
        const float bv = bias[n];
        #pragma unroll
        for (int i = 0; i < 4; ++i)
            #pragma unroll
            for (int r = 0; r < 8; ++r) {
                const int m = m0 + i * 16 + r + 8 * g;
                out[(size_t)m * D_EMB + n] = acc[i][j][r] + bv;
            }
    }
}

// ---------------------------------------------------------------------------
// Host-side launch (graph-capture safe: all async on `stream`, scratch in d_ws)
// ---------------------------------------------------------------------------
extern "C" void kernel_launch(void* const* d_in, const int* in_sizes, int n_in,
                              void* d_out, int out_size, void* d_ws, size_t ws_size,
                              hipStream_t stream) {
    const float* x     = (const float*)d_in[0];   // [2,2048,1024]
    const float* W_qkv = (const float*)d_in[1];   // [1024,3072]
    const float* b_qkv = (const float*)d_in[2];   // [3072]
    const float* W_out = (const float*)d_in[3];   // [1024,1024]
    const float* b_out = (const float*)d_in[4];   // [1024]
    float* out = (float*)d_out;                   // [2,2048,1024] f32

    char* ws = (char*)d_ws;
    _Float16* Xh    = (_Float16*)ws;  ws += (size_t)MROWS * D_EMB * 2;      // 8 MB
    _Float16* WqkvT = (_Float16*)ws;  ws += (size_t)D_QKV * D_EMB * 2;      // 6 MB
    _Float16* WoutT = (_Float16*)ws;  ws += (size_t)D_EMB * D_EMB * 2;      // 2 MB
    _Float16* Qh    = (_Float16*)ws;  ws += (size_t)MROWS * D_EMB * 2;      // 8 MB
    _Float16* Kh    = (_Float16*)ws;  ws += (size_t)MROWS * D_EMB * 2;      // 8 MB
    _Float16* Vt    = (_Float16*)ws;  ws += (size_t)MROWS * D_EMB * 2;      // 8 MB
    _Float16* attnH = (_Float16*)ws;  ws += (size_t)MROWS * D_EMB * 2;      // 8 MB

    // Pre-passes: X -> f16; weights -> transposed [N,K] f16
    {
        int n = MROWS * D_EMB;
        cvt_f32_f16<<<(n + 255) / 256, 256, 0, stream>>>(x, Xh, n);
    }
    {
        int n = D_EMB * D_QKV;
        transpose_f32_f16<<<(n + 255) / 256, 256, 0, stream>>>(W_qkv, WqkvT, D_EMB, D_QKV);
    }
    {
        int n = D_EMB * D_EMB;
        transpose_f32_f16<<<(n + 255) / 256, 256, 0, stream>>>(W_out, WoutT, D_EMB, D_EMB);
    }

    // QKV projection: 64x64 tile per wave
    qkv_gemm<<<dim3(MROWS / 64, D_QKV / 64), 32, 0, stream>>>(Xh, WqkvT, b_qkv,
                                                              Qh, Kh, Vt);

    // Flash attention: (32-query tile) x (batch*head)
    attn_fwd<<<dim3(SEQ / 32, NBATCH * N_HEADS), 32, 0, stream>>>(Qh, Kh, Vt, attnH);

    // Output projection
    out_gemm<<<dim3(MROWS / 64, D_EMB / 64), 32, 0, stream>>>(attnH, WoutT, b_out, out);
}
